// MoeLayer_5600637354144
// MI455X (gfx1250) — matrix operand order repaired and checked by reading, not compiled, not run on hardware
//
#include <hip/hip_runtime.h>
#include <hip/hip_bf16.h>

// Problem constants (match reference)
#define T_TOK 4096
#define DIM   1024
#define HID   2816
#define NEXP  8
#define TOPK  2

typedef __attribute__((ext_vector_type(16))) __bf16 v16bf;
typedef __attribute__((ext_vector_type(8)))  float  v8f;

union ABf { uint4 q[2]; v16bf v; };

__device__ __forceinline__ unsigned int f32_to_bf16_bits(float f) {
  unsigned int u = __float_as_uint(f);
  return (u + 0x7FFFu + ((u >> 16) & 1u)) >> 16;   // round-to-nearest-even
}
__device__ __forceinline__ unsigned int pack_bf16(float lo, float hi) {
  return (f32_to_bf16_bits(hi) << 16) | (f32_to_bf16_bits(lo) & 0xFFFFu);
}

// ---------------------------------------------------------------------------
// x [T,D] fp32 -> bf16 row-major (A-fragment friendly: 16B-aligned K runs)
// ---------------------------------------------------------------------------
__global__ void k_cvt_x(const float* __restrict__ x, unsigned int* __restrict__ xb,
                        int n_uint) {
  int u = blockIdx.x * 256 + threadIdx.x;
  if (u < n_uint) xb[u] = pack_bf16(x[2 * u], x[2 * u + 1]);
}

// ---------------------------------------------------------------------------
// Weights [E,Kdim,Ndim] fp32 -> bf16 tiled in WMMA B-fragment lane layout.
// Tile (e,kt,nt) = 32x16 (KxN). Contiguous K order used here: lane l (col = l&15, g = l>>4),
// dword j holds K = kt*32 + 16*g + 2j, 2j+1.  Storage: lane-major, 8 dwords
// per lane, 256 dwords per tile, tiles ordered (e, nt, kt).
// ---------------------------------------------------------------------------
__global__ void k_cvt_w_tiled(const float* __restrict__ src,
                              unsigned int* __restrict__ dst,
                              int Kdim, int Ndim) {
  int u    = blockIdx.x * 256 + threadIdx.x;   // one dword per thread
  int tile = u >> 8;
  int idx  = u & 255;
  int lane = idx >> 3;
  int j    = idx & 7;
  int ktiles = Kdim >> 5;
  int tpe    = ktiles * (Ndim >> 4);
  int e    = tile / tpe;
  int rem  = tile - e * tpe;
  int nt   = rem / ktiles;
  int kt   = rem - nt * ktiles;
  int k    = kt * 32 + ((lane >> 4) << 4) + 2 * j;
  int n    = nt * 16 + (lane & 15);
  size_t s = ((size_t)e * Kdim + k) * (size_t)Ndim + n;
  dst[u] = pack_bf16(src[s], src[s + Ndim]);
}

// ---------------------------------------------------------------------------
// Gate + top-2 routing. One wave per token.
// ---------------------------------------------------------------------------
__global__ void k_gate(const float* __restrict__ x, const float* __restrict__ gw,
                       int* __restrict__ cnt, int* __restrict__ list,
                       float* __restrict__ cw) {
  int lane = threadIdx.x & 31;
  int wv   = threadIdx.x >> 5;
  int t    = blockIdx.x * 8 + wv;

  float acc[NEXP];
#pragma unroll
  for (int e = 0; e < NEXP; ++e) acc[e] = 0.0f;

  const float* xr = x + (size_t)t * DIM;
  for (int d = lane; d < DIM; d += 32) {
    float xv = xr[d];
#pragma unroll
    for (int e = 0; e < NEXP; ++e) acc[e] += xv * gw[d * NEXP + e];
  }
#pragma unroll
  for (int e = 0; e < NEXP; ++e) {
#pragma unroll
    for (int off = 16; off > 0; off >>= 1) acc[e] += __shfl_xor(acc[e], off, 32);
  }
  if (lane == 0) {
    int i0 = 0; float v0 = acc[0];
#pragma unroll
    for (int e = 1; e < NEXP; ++e) { if (acc[e] > v0) { v0 = acc[e]; i0 = e; } }
    int i1 = -1; float v1 = -3.4e38f;
#pragma unroll
    for (int e = 0; e < NEXP; ++e) {
      if (e != i0 && acc[e] > v1) { v1 = acc[e]; i1 = e; }
    }
    float e1 = __expf(v1 - v0);
    float inv = 1.0f / (1.0f + e1);
    float w0 = inv, w1 = e1 * inv;

    int p0 = atomicAdd(&cnt[i0], 1);
    list[i0 * T_TOK + p0] = t * TOPK + 0;
    cw[t * TOPK + 0] = w0;
    int p1 = atomicAdd(&cnt[i1], 1);
    list[i1 * T_TOK + p1] = t * TOPK + 1;
    cw[t * TOPK + 1] = w1;
  }
}

// ---------------------------------------------------------------------------
// Up-projection: h[slot, :] = silu(x@w1) * (x@w3), bf16 WMMA, per-expert rows.
// Block: 256 threads (8 waves), tile M=64 x N=128. Wave -> mtile (w&3),
// ntile group ((w>>2)*4 .. +3). K-loop over D in 32-chunks, unrolled x2 with
// ping-pong A fragments (A0/A1) so prefetches write the alternate buffer
// directly: no A-register moves, no WMMA->VALU hazard NOPs. Invalid rows
// load from a clamped address; their garbage rows are skipped in the epilogue
// (WMMA output row m depends only on A row m).
// ---------------------------------------------------------------------------
__global__ __launch_bounds__(256) void k_ffn_up(
    const unsigned int* __restrict__ xb, const unsigned int* __restrict__ w1t,
    const unsigned int* __restrict__ w3t, const int* __restrict__ cnt,
    const int* __restrict__ list, unsigned short* __restrict__ hbuf) {
  int e     = blockIdx.z;
  int count = cnt[e];
  int m0    = blockIdx.y * 64;
  if (m0 >= count) return;

  __shared__ int s_slot[64];
  if (threadIdx.x < 64) {
    int r = m0 + (int)threadIdx.x;
    s_slot[threadIdx.x] = (r < count) ? list[e * T_TOK + r] : -1;
  }
  __syncthreads();

  int lane = threadIdx.x & 31;
  int wv   = threadIdx.x >> 5;
  int mt   = wv & 3;
  int nh   = wv >> 2;
  int g    = lane >> 4;
  int ln15 = lane & 15;

  const int ktiles    = DIM / 32;   // 32 (even)
  const int ntilesTot = HID / 16;   // 176
  int ntile0 = blockIdx.x * 8 + nh * 4;

  int aslot = s_slot[mt * 16 + ln15];
  size_t arow = (size_t)((aslot >= 0 ? aslot : 0) >> 1) * DIM;

  const uint4* xb4 = (const uint4*)xb;
  const uint4* w14 = (const uint4*)w1t;
  const uint4* w34 = (const uint4*)w3t;

  size_t abase = (arow + (size_t)(16 * g)) >> 3;   // uint4 units
  size_t tbase = (((size_t)e * ktiles * ntilesTot + (size_t)ntile0 * ktiles) << 6)
                 + (size_t)lane * 2;

  v8f acc1[4], acc3[4];
#pragma unroll
  for (int q = 0; q < 4; ++q) { acc1[q] = (v8f){}; acc3[q] = (v8f){}; }

  ABf A0, A1;
  A0.q[0] = xb4[abase];
  A0.q[1] = xb4[abase + 1];

  for (int kt = 0; kt < ktiles; kt += 2) {
    ABf B1v[4], B3v[4];

    // ---- even half: consumes A0, prefetches A1 (kt+1 < ktiles always) ----
#pragma unroll
    for (int q = 0; q < 4; ++q) {
      size_t tb = tbase + ((size_t)(q * ktiles + kt) << 6);
      B1v[q].q[0] = w14[tb];
      B1v[q].q[1] = w14[tb + 1];
      B3v[q].q[0] = w34[tb];
      B3v[q].q[1] = w34[tb + 1];
    }
    {
      size_t ab = abase + (size_t)((kt + 1) * 4);   // 32 bf16 = 4 uint4
      A1.q[0] = xb4[ab];
      A1.q[1] = xb4[ab + 1];
    }
#pragma unroll
    for (int q = 0; q < 4; ++q) {
      acc1[q] = __builtin_amdgcn_wmma_f32_16x16x32_bf16(
          false, A0.v, false, B1v[q].v, (short)0, acc1[q], false, false);
      acc3[q] = __builtin_amdgcn_wmma_f32_16x16x32_bf16(
          false, A0.v, false, B3v[q].v, (short)0, acc3[q], false, false);
    }

    // ---- odd half: consumes A1, prefetches A0 for next pair ----
#pragma unroll
    for (int q = 0; q < 4; ++q) {
      size_t tb = tbase + ((size_t)(q * ktiles + kt + 1) << 6);
      B1v[q].q[0] = w14[tb];
      B1v[q].q[1] = w14[tb + 1];
      B3v[q].q[0] = w34[tb];
      B3v[q].q[1] = w34[tb + 1];
    }
    if (kt + 2 < ktiles) {
      size_t ab = abase + (size_t)((kt + 2) * 4);
      A0.q[0] = xb4[ab];
      A0.q[1] = xb4[ab + 1];
    }
#pragma unroll
    for (int q = 0; q < 4; ++q) {
      acc1[q] = __builtin_amdgcn_wmma_f32_16x16x32_bf16(
          false, A1.v, false, B1v[q].v, (short)0, acc1[q], false, false);
      acc3[q] = __builtin_amdgcn_wmma_f32_16x16x32_bf16(
          false, A1.v, false, B3v[q].v, (short)0, acc3[q], false, false);
    }
  }

  // Epilogue: h = silu(a1) * a3 -> bf16 scatter to h rows (slot-indexed).
#pragma unroll
  for (int q = 0; q < 4; ++q) {
    int col = (ntile0 + q) * 16 + ln15;
#pragma unroll
    for (int i = 0; i < 8; ++i) {
      int slot = s_slot[mt * 16 + i + 8 * g];
      if (slot >= 0) {
        float a1 = acc1[q][i];
        float hv = (a1 / (1.0f + __expf(-a1))) * acc3[q][i];
        hbuf[(size_t)slot * HID + col] = (unsigned short)f32_to_bf16_bits(hv);
      }
    }
  }
}

// ---------------------------------------------------------------------------
// Down-projection: out[t,:] += cw[slot] * (h[slot,:] @ w2[e]).
// Same ping-pong A; additionally separate B buffers per unrolled half
// (register budget allows it: 32 accum vs 64), removing the WAR dependence
// between the odd half's loads and the even half's WMMAs.
// ---------------------------------------------------------------------------
__global__ __launch_bounds__(256) void k_ffn_down(
    const unsigned int* __restrict__ hb, const unsigned int* __restrict__ w2t,
    const int* __restrict__ cnt, const int* __restrict__ list,
    const float* __restrict__ cw, float* __restrict__ out) {
  int e     = blockIdx.z;
  int count = cnt[e];
  int m0    = blockIdx.y * 64;
  if (m0 >= count) return;

  __shared__ int s_slot[64];
  if (threadIdx.x < 64) {
    int r = m0 + (int)threadIdx.x;
    s_slot[threadIdx.x] = (r < count) ? list[e * T_TOK + r] : -1;
  }
  __syncthreads();

  int lane = threadIdx.x & 31;
  int wv   = threadIdx.x >> 5;
  int mt   = wv & 3;
  int nh   = wv >> 2;
  int g    = lane >> 4;
  int ln15 = lane & 15;

  const int ktiles    = HID / 32;   // 88 (even)
  const int ntilesTot = DIM / 16;   // 64
  int ntile0 = blockIdx.x * 8 + nh * 4;

  int aslot = s_slot[mt * 16 + ln15];
  size_t arow = (size_t)(aslot >= 0 ? aslot : 0) * HID;

  const uint4* hb4 = (const uint4*)hb;
  const uint4* w24 = (const uint4*)w2t;

  size_t abase = (arow + (size_t)(16 * g)) >> 3;
  size_t tbase = (((size_t)e * ktiles * ntilesTot + (size_t)ntile0 * ktiles) << 6)
                 + (size_t)lane * 2;

  v8f acc[4];
#pragma unroll
  for (int q = 0; q < 4; ++q) acc[q] = (v8f){};

  ABf A0, A1;
  A0.q[0] = hb4[abase];
  A0.q[1] = hb4[abase + 1];

  // Pre-load even-half B for kt=0.
  ABf Be[4], Bo[4];
#pragma unroll
  for (int q = 0; q < 4; ++q) {
    size_t tb = tbase + ((size_t)(q * ktiles) << 6);
    Be[q].q[0] = w24[tb];
    Be[q].q[1] = w24[tb + 1];
  }

  for (int kt = 0; kt < ktiles; kt += 2) {
    // Prefetch odd-half B and A1 before consuming even half.
#pragma unroll
    for (int q = 0; q < 4; ++q) {
      size_t tb = tbase + ((size_t)(q * ktiles + kt + 1) << 6);
      Bo[q].q[0] = w24[tb];
      Bo[q].q[1] = w24[tb + 1];
    }
    {
      size_t ab = abase + (size_t)((kt + 1) * 4);
      A1.q[0] = hb4[ab];
      A1.q[1] = hb4[ab + 1];
    }
#pragma unroll
    for (int q = 0; q < 4; ++q) {
      acc[q] = __builtin_amdgcn_wmma_f32_16x16x32_bf16(
          false, A0.v, false, Be[q].v, (short)0, acc[q], false, false);
    }

    // Prefetch next even-half B and A0 before consuming odd half.
    if (kt + 2 < ktiles) {
#pragma unroll
      for (int q = 0; q < 4; ++q) {
        size_t tb = tbase + ((size_t)(q * ktiles + kt + 2) << 6);
        Be[q].q[0] = w24[tb];
        Be[q].q[1] = w24[tb + 1];
      }
      size_t ab = abase + (size_t)((kt + 2) * 4);
      A0.q[0] = hb4[ab];
      A0.q[1] = hb4[ab + 1];
    }
#pragma unroll
    for (int q = 0; q < 4; ++q) {
      acc[q] = __builtin_amdgcn_wmma_f32_16x16x32_bf16(
          false, A1.v, false, Bo[q].v, (short)0, acc[q], false, false);
    }
  }

  // Epilogue: weighted scatter-add into out[token, :] (2 adds/elem total).
#pragma unroll
  for (int q = 0; q < 4; ++q) {
    int col = (ntile0 + q) * 16 + ln15;
#pragma unroll
    for (int i = 0; i < 8; ++i) {
      int slot = s_slot[mt * 16 + i + 8 * g];
      if (slot >= 0) {
        int t   = slot >> 1;
        float w = cw[slot];
        atomicAdd(&out[(size_t)t * DIM + col], w * acc[q][i]);
      }
    }
  }
}

// ---------------------------------------------------------------------------
extern "C" void kernel_launch(void* const* d_in, const int* in_sizes, int n_in,
                              void* d_out, int out_size, void* d_ws, size_t ws_size,
                              hipStream_t stream) {
  (void)in_sizes; (void)n_in; (void)out_size; (void)ws_size;
  const float* x      = (const float*)d_in[0];
  const float* gate_w = (const float*)d_in[1];
  const float* w1     = (const float*)d_in[2];
  const float* w2     = (const float*)d_in[3];
  const float* w3     = (const float*)d_in[4];
  float* out          = (float*)d_out;

  // Workspace carve-out (~193 MB total)
  char* ws = (char*)d_ws;
  size_t off = 0;
  auto carve = [&](size_t bytes) -> void* {
    void* p = ws + off;
    off += (bytes + 255) & ~(size_t)255;
    return p;
  };
  unsigned int*   w1t  = (unsigned int*)  carve((size_t)NEXP * DIM * HID * 2);
  unsigned int*   w3t  = (unsigned int*)  carve((size_t)NEXP * DIM * HID * 2);
  unsigned int*   w2t  = (unsigned int*)  carve((size_t)NEXP * HID * DIM * 2);
  unsigned int*   xb   = (unsigned int*)  carve((size_t)T_TOK * DIM * 2);
  unsigned short* hbuf = (unsigned short*)carve((size_t)T_TOK * TOPK * HID * 2);
  int*            cnt  = (int*)           carve((size_t)NEXP * 4);
  int*            list = (int*)           carve((size_t)NEXP * T_TOK * 4);
  float*          cw   = (float*)         carve((size_t)T_TOK * TOPK * 4);

  hipMemsetAsync(cnt, 0, NEXP * sizeof(int), stream);
  hipMemsetAsync(out, 0, (size_t)T_TOK * DIM * sizeof(float), stream);

  // Precision conversion / weight swizzle (one-shot streaming passes)
  int xu = T_TOK * DIM / 2;
  k_cvt_x<<<xu / 256, 256, 0, stream>>>(x, xb, xu);
  int wu = NEXP * DIM * HID / 2;  // same element count for all three tensors
  k_cvt_w_tiled<<<wu / 256, 256, 0, stream>>>(w1, w1t, DIM, HID);
  k_cvt_w_tiled<<<wu / 256, 256, 0, stream>>>(w3, w3t, DIM, HID);
  k_cvt_w_tiled<<<wu / 256, 256, 0, stream>>>(w2, w2t, HID, DIM);

  // Routing
  k_gate<<<T_TOK / 8, 256, 0, stream>>>(x, gate_w, cnt, list, cw);

  // Expert GEMMs (worst-case grid; blocks past per-expert count early-exit)
  dim3 g1(HID / 128, T_TOK / 64, NEXP);
  k_ffn_up<<<g1, 256, 0, stream>>>(xb, w1t, w3t, cnt, list, hbuf);
  dim3 g2(DIM / 128, T_TOK / 64, NEXP);
  k_ffn_down<<<g2, 256, 0, stream>>>((const unsigned int*)hbuf, w2t, cnt, list, cw, out);
}
